// HolographicAttention_27891517620449
// MI455X (gfx1250) — matrix operand order repaired and checked by reading, not compiled, not run on hardware
//
#include <hip/hip_runtime.h>
#include <hip/hip_bf16.h>
#include <cstdint>
#include <cstddef>

// Problem constants (fixed by the reference)
#define DIMD   1024
#define SEQB   2048      // B*L = 2*1024 rows
#define HEADS  16
#define HD     64

typedef __attribute__((ext_vector_type(16))) __bf16 v16bf;
typedef __attribute__((ext_vector_type(8)))  float  v8f;

__device__ __forceinline__ unsigned short f2bf(float x) {
  unsigned u = __float_as_uint(x);
  u += 0x7FFFu + ((u >> 16) & 1u);          // round-to-nearest-even
  return (unsigned short)(u >> 16);
}

union FragU { v16bf v; int4 q[2]; };

// A-matrix fragment (16x32 bf16, row-major source):
// lane l -> row (l&15); K = (l>>4)*8 + {0..7} in halves 0-7, +16 in halves 8-15
__device__ __forceinline__ v16bf ld_fragA(const unsigned short* p, int stride) {
  const int l = threadIdx.x & 31;
  const unsigned short* q = p + (size_t)(l & 15) * stride + ((l >> 4) << 3);
  FragU u;
  u.q[0] = *(const int4*)(q);
  u.q[1] = *(const int4*)(q + 16);
  return u.v;
}

// B-matrix fragment (32x16 bf16) loaded from row-major rows == B columns:
// lane l -> source row (l&15); K = (l>>4)*8 + {0..7} in halves 0-7, +16 in halves 8-15
__device__ __forceinline__ v16bf ld_fragB(const unsigned short* p, int stride) {
  const int l = threadIdx.x & 31;
  const unsigned short* q = p + (size_t)(l & 15) * stride + ((l >> 4) << 3);
  FragU u;
  u.q[0] = *(const int4*)(q);
  u.q[1] = *(const int4*)(q + 16);
  return u.v;
}

#define WMMA_BF16(a, b, c) \
  __builtin_amdgcn_wmma_f32_16x16x32_bf16(false, (a), false, (b), (short)0, (c), false, false)

// ---------------------------------------------------------------------------
// fp32 -> bf16 conversion
__global__ void __launch_bounds__(256) cvt_bf16_kernel(
    const float* __restrict__ src, unsigned short* __restrict__ dst, int n) {
  int i = blockIdx.x * 256 + threadIdx.x;
  if (i < n) dst[i] = f2bf(src[i]);
}

// ---------------------------------------------------------------------------
// C(MxN) = A(MxK) @ W(NxK)^T + bias ; A,W bf16 row-major.
// One wave computes a 16x64 strip (A fragment reused across 4 N-tiles).
template<bool BF16OUT>
__global__ void __launch_bounds__(256) gemm_xwT_kernel(
    const unsigned short* __restrict__ A,
    const unsigned short* __restrict__ W,
    const float* __restrict__ bias,
    void* __restrict__ Cout,
    int M, int N, int K) {
  const int wid    = (blockIdx.x * 256 + threadIdx.x) >> 5;
  const int ntiles = N >> 6;                       // N/64
  const int mt     = wid / ntiles;
  const int n0     = (wid - mt * ntiles) << 6;     // *64
  if (mt * 16 >= M) return;

  const unsigned short* arow = A + (size_t)mt * 16 * K;
  const unsigned short* w0 = W + (size_t)(n0 +  0) * K;
  const unsigned short* w1 = W + (size_t)(n0 + 16) * K;
  const unsigned short* w2 = W + (size_t)(n0 + 32) * K;
  const unsigned short* w3 = W + (size_t)(n0 + 48) * K;

  v8f c0 = {}, c1 = {}, c2 = {}, c3 = {};
  for (int k0 = 0; k0 < K; k0 += 32) {
    v16bf a  = ld_fragA(arow + k0, K);
    v16bf b0 = ld_fragB(w0 + k0, K);
    v16bf b1 = ld_fragB(w1 + k0, K);
    v16bf b2 = ld_fragB(w2 + k0, K);
    v16bf b3 = ld_fragB(w3 + k0, K);
    c0 = WMMA_BF16(a, b0, c0);
    c1 = WMMA_BF16(a, b1, c1);
    c2 = WMMA_BF16(a, b2, c2);
    c3 = WMMA_BF16(a, b3, c3);
  }

  const int l  = threadIdx.x & 31;
  const int m  = mt * 16 + ((l >> 4) << 3);        // C layout: rows m..m+7 in VGPR 0..7
  const int nn = l & 15;
#pragma unroll
  for (int t = 0; t < 4; ++t) {
    v8f c = (t == 0) ? c0 : (t == 1) ? c1 : (t == 2) ? c2 : c3;
    const int n  = n0 + (t << 4) + nn;
    const float bv = bias[n];
    if (BF16OUT) {
      unsigned short* C = (unsigned short*)Cout;
#pragma unroll
      for (int r = 0; r < 8; ++r) C[(size_t)(m + r) * N + n] = f2bf(c[r] + bv);
    } else {
      float* C = (float*)Cout;
#pragma unroll
      for (int r = 0; r < 8; ++r) C[(size_t)(m + r) * N + n] = c[r] + bv;
    }
  }
}

// ---------------------------------------------------------------------------
// Transpose V (SEQB x DIMD, head-interleaved cols) -> Vt[(b*16+h)*64+d][j]
__global__ void __launch_bounds__(256) transpose_v_kernel(
    const unsigned short* __restrict__ Vr, unsigned short* __restrict__ Vt) {
  int i = blockIdx.x * 256 + threadIdx.x;          // over SEQB*DIMD
  int rm = i >> 10;            // source row  (b*1024 + j)
  int cn = i & 1023;           // source col  (h*64 + d)
  int b = rm >> 10, j = rm & 1023;
  int h = cn >> 6,  d = cn & 63;
  Vt[(size_t)(((b << 4) + h) * 64 + d) * 1024 + j] = Vr[i];
}

// ---------------------------------------------------------------------------
// Fused holographic attention: per wave one (b, h, 16-row) tile.
// Pass 1: sim strip via WMMA, score/sigmoid in registers, bf16 w -> LDS,
//         per-lane partial rowsums. Pass 2: (w @ v) via WMMA from LDS/Vt,
//         output scaled by 1/rowsum.
__global__ void __launch_bounds__(64) attn_kernel(
    const unsigned short* __restrict__ Q,
    const unsigned short* __restrict__ Kmat,
    const unsigned short* __restrict__ Vt,
    unsigned short* __restrict__ AO,
    const float* __restrict__ alpha,
    const float* __restrict__ phi,
    const float* __restrict__ temperature) {
  __shared__ unsigned short wlds[2 * 16 * 1024];   // 64 KB: two waves/block
  const int lane = threadIdx.x & 31;
  const int wib  = threadIdx.x >> 5;
  const int wid  = blockIdx.x * 2 + wib;           // 0..2047
  const int it = wid & 63;
  const int h  = (wid >> 6) & 15;
  const int b  = wid >> 10;
  unsigned short* wbuf = wlds + wib * (16 * 1024);

  const float temp   = fabsf(temperature[0]) + 0.1f;
  const float sscale = 1.0f / (8.0f * temp);       // 1/sqrt(64)/temp
  const float TWO_PI = 6.28318530717958647692f;
  const float kf[5] = {TWO_PI / 31.0f, TWO_PI / 37.0f, TWO_PI / 41.0f,
                       TWO_PI / 43.0f, TWO_PI / 47.0f};
  float al[5], ph[5];
#pragma unroll
  for (int f = 0; f < 5; ++f) { al[f] = alpha[h * 5 + f]; ph[f] = phi[h * 5 + f]; }

  const int i0 = it << 4;
  const unsigned short* qbase = Q + (size_t)(b * 1024 + i0) * 1024 + h * 64;
  v16bf aq0 = ld_fragA(qbase,      1024);          // K-dim (hd) chunk 0..31
  v16bf aq1 = ld_fragA(qbase + 32, 1024);          // chunk 32..63

  const int mrow = (lane >> 4) << 3;               // C-layout row group
  const int nn   = lane & 15;
  float rs[8] = {0.f, 0.f, 0.f, 0.f, 0.f, 0.f, 0.f, 0.f};

  for (int jt = 0; jt < 64; ++jt) {
    const unsigned short* kbase = Kmat + (size_t)(b * 1024 + (jt << 4)) * 1024 + h * 64;
    v16bf bk0 = ld_fragB(kbase,      1024);
    v16bf bk1 = ld_fragB(kbase + 32, 1024);
    v8f c = {};
    c = WMMA_BF16(aq0, bk0, c);
    c = WMMA_BF16(aq1, bk1, c);
#pragma unroll
    for (int r = 0; r < 8; ++r) {
      float sim = c[r] * sscale;
      float sc = 0.f;
#pragma unroll
      for (int f = 0; f < 5; ++f) sc += al[f] * __sinf(sim * kf[f] + ph[f]);
      float w = 1.0f / (1.0f + __expf(-sc));       // sigmoid
      rs[r] += w;
      wbuf[(mrow + r) * 1024 + (jt << 4) + nn] = f2bf(w);
    }
  }

  // Row sums live spread across the 16 lanes of each half-wave -> xor reduce.
#pragma unroll
  for (int r = 0; r < 8; ++r) {
    float s = rs[r];
    s += __shfl_xor(s, 1, 32);
    s += __shfl_xor(s, 2, 32);
    s += __shfl_xor(s, 4, 32);
    s += __shfl_xor(s, 8, 32);
    rs[r] = 1.0f / (s + 1e-10f);
  }

  // Pass 2: out(16x64) = w(16x1024) @ v(1024x64) ; B fragments from Vt rows.
  const unsigned short* vtb = Vt + (size_t)((b * 16 + h) * 64) * 1024;
  v8f c0 = {}, c1 = {}, c2 = {}, c3 = {};
  for (int kt = 0; kt < 32; ++kt) {
    v16bf aw = ld_fragA(wbuf + (kt << 5), 1024);   // from LDS
    v16bf b0 = ld_fragB(vtb + (size_t) 0 * 1024 + (kt << 5), 1024);
    v16bf b1 = ld_fragB(vtb + (size_t)16 * 1024 + (kt << 5), 1024);
    v16bf b2 = ld_fragB(vtb + (size_t)32 * 1024 + (kt << 5), 1024);
    v16bf b3 = ld_fragB(vtb + (size_t)48 * 1024 + (kt << 5), 1024);
    c0 = WMMA_BF16(aw, b0, c0);
    c1 = WMMA_BF16(aw, b1, c1);
    c2 = WMMA_BF16(aw, b2, c2);
    c3 = WMMA_BF16(aw, b3, c3);
  }
#pragma unroll
  for (int t = 0; t < 4; ++t) {
    v8f c = (t == 0) ? c0 : (t == 1) ? c1 : (t == 2) ? c2 : c3;
    const int d0 = t << 4;
#pragma unroll
    for (int r = 0; r < 8; ++r) {
      float o = c[r] * rs[r];                      // fold 1/rowsum into output
      AO[(size_t)(b * 1024 + i0 + mrow + r) * 1024 + h * 64 + d0 + nn] = f2bf(o);
    }
  }
}

// ---------------------------------------------------------------------------
// LayerNorm over rows of 1024 (biased variance, eps=1e-5)
__global__ void __launch_bounds__(256) ln_kernel(
    const float* __restrict__ X, const float* __restrict__ gamma,
    const float* __restrict__ beta, float* __restrict__ Y) {
  __shared__ float p1[8], p2[8];
  const int row = blockIdx.x;
  const float* x = X + (size_t)row * 1024;
  float v[4], s1 = 0.f, s2 = 0.f;
#pragma unroll
  for (int i = 0; i < 4; ++i) {
    v[i] = x[threadIdx.x + i * 256];
    s1 += v[i];
    s2 += v[i] * v[i];
  }
#pragma unroll
  for (int m = 16; m >= 1; m >>= 1) {
    s1 += __shfl_xor(s1, m, 32);
    s2 += __shfl_xor(s2, m, 32);
  }
  if ((threadIdx.x & 31) == 0) { p1[threadIdx.x >> 5] = s1; p2[threadIdx.x >> 5] = s2; }
  __syncthreads();
  float t1 = 0.f, t2 = 0.f;
#pragma unroll
  for (int i = 0; i < 8; ++i) { t1 += p1[i]; t2 += p2[i]; }
  const float mean = t1 * (1.0f / 1024.0f);
  const float var  = t2 * (1.0f / 1024.0f) - mean * mean;
  const float rstd = rsqrtf(var + 1e-5f);
  float* y = Y + (size_t)row * 1024;
#pragma unroll
  for (int i = 0; i < 4; ++i) {
    const int col = threadIdx.x + i * 256;
    y[col] = (v[i] - mean) * rstd * gamma[col] + beta[col];
  }
}

// ---------------------------------------------------------------------------
extern "C" void kernel_launch(void* const* d_in, const int* in_sizes, int n_in,
                              void* d_out, int out_size, void* d_ws, size_t ws_size,
                              hipStream_t stream) {
  const float* x    = (const float*)d_in[0];
  const float* Wq   = (const float*)d_in[1];
  const float* bq   = (const float*)d_in[2];
  const float* Wk   = (const float*)d_in[3];
  const float* bk   = (const float*)d_in[4];
  const float* Wv   = (const float*)d_in[5];
  const float* bv   = (const float*)d_in[6];
  const float* Wo   = (const float*)d_in[7];
  const float* bo   = (const float*)d_in[8];
  const float* alpha = (const float*)d_in[9];
  const float* phi   = (const float*)d_in[10];
  const float* temp  = (const float*)d_in[11];
  const float* gamma = (const float*)d_in[12];
  const float* beta  = (const float*)d_in[13];

  uint8_t* w8 = (uint8_t*)d_ws;
  size_t off = 0;
  auto take = [&](size_t bytes) -> void* {
    void* p = (void*)(w8 + off);
    off += (bytes + 255) & ~(size_t)255;
    return p;
  };
  unsigned short* Xb  = (unsigned short*)take((size_t)SEQB * DIMD * 2);
  unsigned short* Wqb = (unsigned short*)take((size_t)DIMD * DIMD * 2);
  unsigned short* Wkb = (unsigned short*)take((size_t)DIMD * DIMD * 2);
  unsigned short* Wvb = (unsigned short*)take((size_t)DIMD * DIMD * 2);
  unsigned short* Wob = (unsigned short*)take((size_t)DIMD * DIMD * 2);
  unsigned short* Qb  = (unsigned short*)take((size_t)SEQB * DIMD * 2);
  unsigned short* Kb  = (unsigned short*)take((size_t)SEQB * DIMD * 2);
  unsigned short* Vb  = (unsigned short*)take((size_t)SEQB * DIMD * 2);
  unsigned short* Vt  = (unsigned short*)take((size_t)SEQB * DIMD * 2);
  unsigned short* AOb = (unsigned short*)take((size_t)SEQB * DIMD * 2);
  float*          Ofp = (float*)take((size_t)SEQB * DIMD * 4);

  const int nX = SEQB * DIMD;       // 2,097,152
  const int nW = DIMD * DIMD;       // 1,048,576
  cvt_bf16_kernel<<<(nX + 255) / 256, 256, 0, stream>>>(x,  Xb,  nX);
  cvt_bf16_kernel<<<(nW + 255) / 256, 256, 0, stream>>>(Wq, Wqb, nW);
  cvt_bf16_kernel<<<(nW + 255) / 256, 256, 0, stream>>>(Wk, Wkb, nW);
  cvt_bf16_kernel<<<(nW + 255) / 256, 256, 0, stream>>>(Wv, Wvb, nW);
  cvt_bf16_kernel<<<(nW + 255) / 256, 256, 0, stream>>>(Wo, Wob, nW);

  // (M/16)*(N/64) = 128*16 = 2048 waves -> 256 blocks of 8 waves
  const int gemm_blocks = (SEQB / 16) * (DIMD / 64) / 8;
  gemm_xwT_kernel<true><<<gemm_blocks, 256, 0, stream>>>(Xb, Wqb, bq, (void*)Qb, SEQB, DIMD, DIMD);
  gemm_xwT_kernel<true><<<gemm_blocks, 256, 0, stream>>>(Xb, Wkb, bk, (void*)Kb, SEQB, DIMD, DIMD);
  gemm_xwT_kernel<true><<<gemm_blocks, 256, 0, stream>>>(Xb, Wvb, bv, (void*)Vb, SEQB, DIMD, DIMD);

  transpose_v_kernel<<<(nX + 255) / 256, 256, 0, stream>>>(Vb, Vt);

  // 2048 (b,h,row-tile) waves, 2 waves per block
  attn_kernel<<<1024, 64, 0, stream>>>(Qb, Kb, Vt, AOb, alpha, phi, temp);

  gemm_xwT_kernel<false><<<gemm_blocks, 256, 0, stream>>>(AOb, Wob, bo, (void*)Ofp, SEQB, DIMD, DIMD);

  ln_kernel<<<SEQB, 256, 0, stream>>>(Ofp, gamma, beta, (float*)d_out);
}
